// RPETableAttention_76914274336999
// MI455X (gfx1250) — matrix operand order repaired and checked by reading, not compiled, not run on hardware
//
#include <hip/hip_runtime.h>
#include <hip/hip_bf16.h>

typedef __attribute__((ext_vector_type(16))) _Float16 v16h;
typedef __attribute__((ext_vector_type(8)))  _Float16 v8h;
typedef __attribute__((ext_vector_type(4)))  _Float16 v4h;
typedef __attribute__((ext_vector_type(8)))  float    v8f;

// ---------------------------------------------------------------------------
// WMMA helpers (CDNA5 gfx1250, wave32).
// A 16x32 f16: lane<16 -> row=lane, halves[0..7]=K0..7, halves[8..15]=K16..23
//              lane>=16 -> row=lane-16, halves=K8..15 / K24..31
// B 32x16 f16: N=lane%16; lane<16 -> halves=K0..7 / K16..23, lane>=16 -> K8..15 / K24..31
// C/D 16x16 f32: VGPR r -> row r + (lane<16?0:8), col lane%16
// ---------------------------------------------------------------------------
__device__ __forceinline__ v8f wmma16(v16h a, v16h b, v8f c) {
    return __builtin_amdgcn_wmma_f32_16x16x32_f16(false, a, false, b,
                                                  (short)0, c, false, false);
}

__device__ __forceinline__ v16h load_a16(const _Float16* base, int strideH, int lane) {
    int row  = lane & 15;
    int koff = (lane & 16) ? 8 : 0;
    const _Float16* p = base + row * strideH + koff;
    v8h lo = *(const v8h*)(p);
    v8h hi = *(const v8h*)(p + 16);
    v16h r;
#pragma unroll
    for (int i = 0; i < 8; ++i) { r[i] = lo[i]; r[8 + i] = hi[i]; }
    return r;
}

__device__ __forceinline__ v16h load_b16(const _Float16* base, int strideH, int lane) {
    int n    = lane & 15;
    int koff = (lane & 16) ? 8 : 0;
    const _Float16* p = base + n * strideH + koff;
    v8h lo = *(const v8h*)(p);
    v8h hi = *(const v8h*)(p + 16);
    v16h r;
#pragma unroll
    for (int i = 0; i < 8; ++i) { r[i] = lo[i]; r[8 + i] = hi[i]; }
    return r;
}

// ---------------------------------------------------------------------------
// CDNA5 async global->LDS copies (ASYNCcnt path, no VGPR round-trip).
// vdst = per-lane LDS byte address (low 32 bits of generic shared pointer).
// ---------------------------------------------------------------------------
__device__ __forceinline__ void async_copy_b128(void* lds, const void* g) {
    unsigned l = (unsigned)(uintptr_t)lds;
    unsigned long long ga = (unsigned long long)(uintptr_t)g;
    asm volatile("global_load_async_to_lds_b128 %0, %1, off"
                 :: "v"(l), "v"(ga) : "memory");
}
__device__ __forceinline__ void async_copy_b32(void* lds, const void* g) {
    unsigned l = (unsigned)(uintptr_t)lds;
    unsigned long long ga = (unsigned long long)(uintptr_t)g;
    asm volatile("global_load_async_to_lds_b32 %0, %1, off"
                 :: "v"(l), "v"(ga) : "memory");
}
__device__ __forceinline__ void wait_async0() {
    asm volatile("s_wait_asynccnt 0x0" ::: "memory");
}

// ---------------------------------------------------------------------------
// Constants
// ---------------------------------------------------------------------------
#define BATCH 256
#define NTOK  256
#define DIM   384
#define HEADS 12
#define HD    32
#define TABLE 961
#define QSCALE 0.17677669529663687f   // 1/sqrt(32)

// ---------------------------------------------------------------------------
// Kernel 0: weight conversion f32 -> f16
// ---------------------------------------------------------------------------
__global__ void __launch_bounds__(256)
cvt_weights(const float* __restrict__ wq, const float* __restrict__ wp,
            _Float16* __restrict__ wqh, _Float16* __restrict__ wph) {
    int i = blockIdx.x * 256 + threadIdx.x;
    const int NQ = 3 * DIM * DIM;
    const int NP = DIM * DIM;
    if (i < NQ) wqh[i] = (_Float16)wq[i];
    if (i < NP) wph[i] = (_Float16)wp[i];
}

// ---------------------------------------------------------------------------
// Kernel 1: QKV projection.  y = x @ w_qkv^T -> head-major f16 Q/K/V,
// Q pre-scaled.  M=65536, N=1152, K=384.  128x64 tile, 8 waves, 2x2 WMMA.
// A (f32->f16) staged via VALU; B (f16) staged via async-to-LDS.
// ---------------------------------------------------------------------------
__global__ void __launch_bounds__(256)
qkv_gemm(const float* __restrict__ x, const _Float16* __restrict__ Wh,
         _Float16* __restrict__ Q, _Float16* __restrict__ K,
         _Float16* __restrict__ V) {
    __shared__ _Float16 As[128 * 40];
    __shared__ _Float16 Bs[64 * 40];

    const int t = threadIdx.x, lane = t & 31, wave = t >> 5;
    const int m0  = blockIdx.x * 128;
    const int jj0 = blockIdx.y * 64;
    const int wr0 = (wave >> 1) * 32;
    const int wc0 = (wave & 1) * 32;

    v8f acc[2][2] = {};

    for (int kb = 0; kb < 12; ++kb) {
        const int k0 = kb * 32;
        // Stage B asynchronously (pure f16 copy)
        {
            int j = t >> 2, kc = (t & 3) * 8;
            async_copy_b128(Bs + j * 40 + kc,
                            Wh + (size_t)(jj0 + j) * DIM + k0 + kc);
        }
        // Stage A: 128x32 f32 -> f16 via VALU
#pragma unroll
        for (int i = 0; i < 4; ++i) {
            int idx4 = t + i * 256;
            int r = idx4 >> 3, c4 = (idx4 & 7) * 4;
            const float* gp = x + (size_t)(m0 + r) * DIM + k0 + c4;
            float4 f = *(const float4*)gp;
            if (kb + 1 < 12) __builtin_prefetch(gp + 32, 0, 0);
            v4h hv; hv[0] = (_Float16)f.x; hv[1] = (_Float16)f.y;
                    hv[2] = (_Float16)f.z; hv[3] = (_Float16)f.w;
            *(v4h*)(As + r * 40 + c4) = hv;
        }
        wait_async0();
        __syncthreads();

        v16h a0 = load_a16(As + (wr0 +  0) * 40, 40, lane);
        v16h a1 = load_a16(As + (wr0 + 16) * 40, 40, lane);
        v16h b0 = load_b16(Bs + (wc0 +  0) * 40, 40, lane);
        v16h b1 = load_b16(Bs + (wc0 + 16) * 40, 40, lane);
        acc[0][0] = wmma16(a0, b0, acc[0][0]);
        acc[0][1] = wmma16(a0, b1, acc[0][1]);
        acc[1][0] = wmma16(a1, b0, acc[1][0]);
        acc[1][1] = wmma16(a1, b1, acc[1][1]);
        __syncthreads();
    }

    // Epilogue: scatter into head-major Q/K/V (no integer division)
#pragma unroll
    for (int sr = 0; sr < 2; ++sr)
#pragma unroll
    for (int sc = 0; sc < 2; ++sc) {
        int colbase = jj0 + wc0 + sc * 16;
        int sel = (colbase >= 2 * DIM) ? 2 : ((colbase >= DIM) ? 1 : 0);
        int c   = colbase - sel * DIM + (lane & 15);
        int h   = c >> 5, d = c & 31;
        _Float16* dst = (sel == 0) ? Q : ((sel == 1) ? K : V);
        float mul = (sel == 0) ? QSCALE : 1.0f;
#pragma unroll
        for (int r = 0; r < 8; ++r) {
            int m  = m0 + wr0 + sr * 16 + r + ((lane & 16) ? 8 : 0);
            int b  = m >> 8, nn = m & 255;
            size_t off = (((size_t)(b * HEADS + h)) * NTOK + nn) * HD + d;
            dst[off] = (_Float16)(acc[sr][sc][r] * mul);
        }
    }
}

// ---------------------------------------------------------------------------
// Kernel 2: attention for one (b,h), 128 query rows per block (8 waves x 16).
// Dynamic LDS layout (bytes):
//   Ks   @ 0       : 256 x 40 halves                                20480
//   Vts  @ 20480   : 32  x 264 halves (transposed)                  16896
//   bias @ 37376   : 961 f32 (column h of rel_table)                 (4096 slot)
//   Sf   @ 41472   : 8 x 16 x 264 f32                              135168
//   Ph   @ 176640  : 8 x 16 x 264 f16                               67584
//   inv  @ 244224  : 128 f32                                          512
// total 244736 B
// ---------------------------------------------------------------------------
#define ATTN_LDS_BYTES 244736

__global__ void __launch_bounds__(256)
attn_kernel(const _Float16* __restrict__ Q, const _Float16* __restrict__ K,
            const _Float16* __restrict__ V, const float* __restrict__ rel,
            _Float16* __restrict__ Octx) {
    extern __shared__ char smem[];
    _Float16* Ks    = (_Float16*)(smem);
    _Float16* Vts   = (_Float16*)(smem + 20480);
    float*    biasL = (float*)   (smem + 37376);
    float*    Sf    = (float*)   (smem + 41472);
    _Float16* Ph    = (_Float16*)(smem + 176640);
    float*    rowInv= (float*)   (smem + 244224);

    const int t = threadIdx.x, lane = t & 31, wave = t >> 5;
    const int wg   = blockIdx.x;
    const int half = wg & 1;
    const int bh   = wg >> 1;              // = b*12 + h
    const int h    = bh % HEADS;
    const int b    = bh / HEADS;

    const size_t bhOff = (size_t)bh * NTOK * HD;
    const _Float16* Qg = Q + bhOff;
    const _Float16* Kg = K + bhOff;
    const _Float16* Vg = V + bhOff;

    // Stage K rows + bias column via async-to-LDS; V transposed via VGPRs
    {
        int n = t;
#pragma unroll
        for (int i = 0; i < 4; ++i)
            async_copy_b128(Ks + n * 40 + i * 8, Kg + n * HD + i * 8);
#pragma unroll
        for (int i = 0; i < 4; ++i) {
            v8h v = *(const v8h*)(Vg + n * HD + i * 8);
#pragma unroll
            for (int e = 0; e < 8; ++e) Vts[(i * 8 + e) * 264 + n] = v[e];
        }
    }
    for (int i = t; i < TABLE; i += 256)
        async_copy_b32(biasL + i, rel + (size_t)i * HEADS + h);
    wait_async0();
    __syncthreads();

    const int qrow0 = half * 128 + wave * 16;
    v16h aQ = load_a16(Qg + (size_t)qrow0 * HD, HD, lane);
    float* Sw = Sf + wave * 16 * 264;

    // S = Q K^T + bias : 16 WMMA per wave
    for (int jt = 0; jt < 16; ++jt) {
        v16h bK = load_b16(Ks + jt * 16 * 40, 40, lane);
        v8f s = {};
        s = wmma16(aQ, bK, s);
        int col = jt * 16 + (lane & 15);
        int yj = col >> 4, xj = col & 15;
#pragma unroll
        for (int r = 0; r < 8; ++r) {
            int ri = r + ((lane & 16) ? 8 : 0);
            int qi = qrow0 + ri;
            int yi = qi >> 4, xi = qi & 15;
            float bias = biasL[(yi - yj + 15) * 31 + (xi - xj + 15)];
            Sw[ri * 264 + col] = s[r] + bias;
        }
    }
    __syncthreads();

    // Softmax: 2 lanes per row (split halves, shfl_xor(16) merge)
    {
        int row = lane & 15;
        int cb  = (lane & 16) ? 128 : 0;
        float* srow = Sw + row * 264;
        _Float16* prow = Ph + (wave * 16 + row) * 264;
        float m = -1e30f;
        for (int c = 0; c < 128; ++c) m = fmaxf(m, srow[cb + c]);
        m = fmaxf(m, __shfl_xor(m, 16, 32));
        float ssum = 0.f;
        for (int c = 0; c < 128; ++c) {
            float e = __expf(srow[cb + c] - m);
            ssum += e;
            prow[cb + c] = (_Float16)e;
        }
        ssum += __shfl_xor(ssum, 16, 32);
        if (lane < 16) rowInv[wave * 16 + row] = 1.0f / ssum;
    }
    __syncthreads();

    // O = P V : 16 WMMA per wave
    const _Float16* Pw = Ph + wave * 16 * 264;
    v8f o0 = {}, o1 = {};
    for (int kk = 0; kk < 8; ++kk) {
        v16h aP  = load_a16(Pw + kk * 32, 264, lane);
        v16h bv0 = load_b16(Vts + 0 * 264 + kk * 32, 264, lane);
        v16h bv1 = load_b16(Vts + 16 * 264 + kk * 32, 264, lane);
        o0 = wmma16(aP, bv0, o0);
        o1 = wmma16(aP, bv1, o1);
    }

    int dcol = lane & 15;
#pragma unroll
    for (int r = 0; r < 8; ++r) {
        int ri = r + ((lane & 16) ? 8 : 0);
        int qi = qrow0 + ri;
        float inv = rowInv[wave * 16 + ri];
        size_t obase = ((size_t)(b * NTOK + qi)) * DIM + h * HD;
        Octx[obase + dcol]      = (_Float16)(o0[r] * inv);
        Octx[obase + 16 + dcol] = (_Float16)(o1[r] * inv);
    }
}

// ---------------------------------------------------------------------------
// Kernel 3: output projection.  out = Octx @ w_proj^T + b_proj (f32 out).
// All-f16 operands -> fully async staging with LDS double buffering:
// tile k+1 streams into the alternate buffer while WMMAs consume tile k.
// ---------------------------------------------------------------------------
__global__ void __launch_bounds__(256)
proj_gemm(const _Float16* __restrict__ O, const _Float16* __restrict__ Wh,
          const float* __restrict__ bproj, float* __restrict__ out) {
    __shared__ _Float16 As[2][128 * 40];
    __shared__ _Float16 Bs[2][64 * 40];

    const int t = threadIdx.x, lane = t & 31, wave = t >> 5;
    const int m0  = blockIdx.x * 128;
    const int jj0 = blockIdx.y * 64;
    const int wr0 = (wave >> 1) * 32;
    const int wc0 = (wave & 1) * 32;

    // per-thread staging coordinates
    const int ra0 = (t + 0)   >> 2, ka0 = ((t + 0)   & 3) * 8;
    const int ra1 = (t + 256) >> 2, ka1 = ((t + 256) & 3) * 8;
    const int jb  = t >> 2,         kbw = (t & 3) * 8;

    auto stage = [&](int buf, int kb) {
        int k0 = kb * 32;
        async_copy_b128(&As[buf][ra0 * 40 + ka0],
                        O + (size_t)(m0 + ra0) * DIM + k0 + ka0);
        async_copy_b128(&As[buf][ra1 * 40 + ka1],
                        O + (size_t)(m0 + ra1) * DIM + k0 + ka1);
        async_copy_b128(&Bs[buf][jb * 40 + kbw],
                        Wh + (size_t)(jj0 + jb) * DIM + k0 + kbw);
    };

    v8f acc[2][2] = {};

    stage(0, 0);
    wait_async0();
    __syncthreads();

    for (int kb = 0; kb < 12; ++kb) {
        const int buf = kb & 1;
        if (kb + 1 < 12) stage(buf ^ 1, kb + 1);

        v16h a0 = load_a16(&As[buf][(wr0 +  0) * 40], 40, lane);
        v16h a1 = load_a16(&As[buf][(wr0 + 16) * 40], 40, lane);
        v16h b0 = load_b16(&Bs[buf][(wc0 +  0) * 40], 40, lane);
        v16h b1 = load_b16(&Bs[buf][(wc0 + 16) * 40], 40, lane);
        acc[0][0] = wmma16(a0, b0, acc[0][0]);
        acc[0][1] = wmma16(a0, b1, acc[0][1]);
        acc[1][0] = wmma16(a1, b0, acc[1][0]);
        acc[1][1] = wmma16(a1, b1, acc[1][1]);

        wait_async0();      // next tile landed
        __syncthreads();    // everyone done reading current tile (dscnt waited)
    }

#pragma unroll
    for (int sr = 0; sr < 2; ++sr)
#pragma unroll
    for (int sc = 0; sc < 2; ++sc) {
        int jj = jj0 + wc0 + sc * 16 + (lane & 15);
        float bias = bproj[jj];
#pragma unroll
        for (int r = 0; r < 8; ++r) {
            int m = m0 + wr0 + sr * 16 + r + ((lane & 16) ? 8 : 0);
            out[(size_t)m * DIM + jj] = acc[sr][sc][r] + bias;
        }
    }
}

// ---------------------------------------------------------------------------
// Host launcher
// ---------------------------------------------------------------------------
extern "C" void kernel_launch(void* const* d_in, const int* in_sizes, int n_in,
                              void* d_out, int out_size, void* d_ws, size_t ws_size,
                              hipStream_t stream) {
    const float* x     = (const float*)d_in[0];
    const float* wqkv  = (const float*)d_in[1];
    const float* rel   = (const float*)d_in[2];
    const float* wproj = (const float*)d_in[3];
    const float* bproj = (const float*)d_in[4];
    float* out = (float*)d_out;

    char* ws = (char*)d_ws;
    _Float16* WqkvH = (_Float16*)(ws + 0);          //  884736
    _Float16* WprojH= (_Float16*)(ws + 884736);     //  294912
    _Float16* Qh    = (_Float16*)(ws + 1179648);    // 50331648
    _Float16* Kh    = (_Float16*)(ws + 51511296);   // 50331648
    _Float16* Vh    = (_Float16*)(ws + 101842944);  // 50331648
    _Float16* Oh    = (_Float16*)(ws + 152174592);  // 50331648  (end 202506240)

    cvt_weights<<<1728, 256, 0, stream>>>(wqkv, wproj, WqkvH, WprojH);
    qkv_gemm<<<dim3(512, 18), 256, 0, stream>>>(x, WqkvH, Qh, Kh, Vh);
    attn_kernel<<<2 * BATCH * HEADS, 256, ATTN_LDS_BYTES, stream>>>(Qh, Kh, Vh, rel, Oh);
    proj_gemm<<<dim3(512, 6), 256, 0, stream>>>(Oh, WprojH, bproj, out);
}
